// SCR_82626580841026
// MI455X (gfx1250) — compile-verified, matrix operand order checked by reading, not run on hardware
//
#include <hip/hip_runtime.h>
#include <hip/hip_bf16.h>
#include <math.h>

// ---------------------------------------------------------------------------
// Problem constants (match reference)
// ---------------------------------------------------------------------------
#define B_  128
#define C_  64
#define H_  1024
#define T_  3
#define E_  6
#define N_  (B_ * C_)      // 8192 tokens

// ---------------------------------------------------------------------------
// WMMA / async-LDS plumbing
// ---------------------------------------------------------------------------
typedef __attribute__((ext_vector_type(16))) __bf16 bf16x16;
typedef __attribute__((ext_vector_type(8)))  float  f32x8;

#if defined(__has_builtin)
#  if __has_builtin(__builtin_amdgcn_global_load_async_to_lds_b128)
#    define ASYNC_LDS 1
#  endif
#  if __has_builtin(__builtin_amdgcn_s_wait_asynccnt)
#    define HAVE_WAIT_ASYNC 1
#  endif
#endif

// Builtin expects: (global int4* src, lds int4* dst, imm offset, imm cpol)
// (param 1 confirmed by compiler diagnostic: vector_size(16) int, AS1)
typedef int i32x4_vs __attribute__((vector_size(16)));
typedef __attribute__((address_space(1))) i32x4_vs* as1_i32x4_p;
typedef __attribute__((address_space(3))) i32x4_vs* as3_i32x4_p;

// 16-byte global(bf16) -> LDS copy; async DMA path on CDNA5, sync fallback.
__device__ __forceinline__ void cp16_g2l(__bf16* l, const __bf16* g) {
#ifdef ASYNC_LDS
    __builtin_amdgcn_global_load_async_to_lds_b128(
        (as1_i32x4_p)g, (as3_i32x4_p)l, 0, 0);
#else
    *reinterpret_cast<uint4*>(l) = *reinterpret_cast<const uint4*>(g);
#endif
}

__device__ __forceinline__ void wait_async0() {
#ifdef ASYNC_LDS
#  ifdef HAVE_WAIT_ASYNC
    __builtin_amdgcn_s_wait_asynccnt(0);
#  else
    asm volatile("s_wait_asynccnt 0" ::: "memory");
#  endif
#endif
}

__device__ __forceinline__ bf16x16 load_frag16(const __bf16* p) {
    // two 16-byte LDS loads -> 8 VGPRs of bf16 (matches 16-bit A/B lane layout)
    union { bf16x16 v; uint4 q[2]; } u;
    u.q[0] = *reinterpret_cast<const uint4*>(p);
    u.q[1] = *reinterpret_cast<const uint4*>(p + 16);
    return u.v;
}

// ---------------------------------------------------------------------------
// bf16 WMMA GEMM:  D = act(A[M,K] * W[K,Nc] + bias), with A in bf16 row-major
// and W pre-transposed+converted to Wt[Nc,K] bf16 (K contiguous -> both LDS
// tiles are plain row copies: async-DMA friendly, no staging transpose).
// 128x128 block tile, 64-wide K stage, double-buffered LDS, 8 waves (4Mx2N),
// each wave owns 2x4 16x16 tiles -> 16 v_wmma per stage.
// ---------------------------------------------------------------------------
#define BM  128
#define BN  128
#define BKS 64      // K elements staged per buffer (2 WMMA k-steps)
#define LDP 72      // LDS row pitch (bf16): 144B -> conflict-free, 16B aligned

__global__ __launch_bounds__(256)
void gemm_bf16_kernel(const __bf16* __restrict__ A,    // [M,K] bf16
                      const __bf16* __restrict__ Wt,   // [Nc,K] bf16
                      const float*  __restrict__ bias, // [Nc]
                      float*        __restrict__ Df,   // optional f32 out
                      __bf16*       __restrict__ Dbf,  // optional bf16 out
                      int M, int Nc, int K, int relu)
{
    __shared__ __align__(16) __bf16 Asb[2][BM * LDP];
    __shared__ __align__(16) __bf16 Bsb[2][BN * LDP];

    const int tid  = threadIdx.x;
    const int lane = tid & 31;
    const int wave = tid >> 5;
    const int wm   = wave & 3;         // wave row (32 rows)
    const int wn   = wave >> 2;        // wave col (64 cols)
    const int bm   = blockIdx.x * BM;
    const int bn   = blockIdx.y * BN;
    const int half = lane >> 4;
    const int l16  = lane & 15;

    f32x8 acc[2][4] = {};

    // stage one 128x64 bf16 tile of A and of Wt into LDS buffer `buf`
    auto stage = [&](int k0, int buf) {
#pragma unroll
        for (int i = 0; i < 4; ++i) {
            int idx = tid + i * 256;       // 0..1023
            int r   = idx >> 3;            // row 0..127
            int c   = (idx & 7) * 8;       // bf16 offset 0..56 (16B chunks)
            cp16_g2l(&Asb[buf][r * LDP + c], A  + (size_t)(bm + r) * K + k0 + c);
            cp16_g2l(&Bsb[buf][r * LDP + c], Wt + (size_t)(bn + r) * K + k0 + c);
        }
    };

    stage(0, 0);
    wait_async0();
    __syncthreads();

    int cur = 0;
    for (int k0 = 0; k0 < K; k0 += BKS) {
        if (k0 + BKS < K) stage(k0 + BKS, cur ^ 1);   // DMA next stage

#pragma unroll
        for (int ks = 0; ks < 2; ++ks) {              // two K=32 WMMA steps
            bf16x16 afrag[2], bfrag[4];
#pragma unroll
            for (int i = 0; i < 2; ++i)
                afrag[i] = load_frag16(
                    &Asb[cur][(wm * 32 + i * 16 + l16) * LDP + ks * 32 + 8 * half]);
#pragma unroll
            for (int j = 0; j < 4; ++j)
                bfrag[j] = load_frag16(
                    &Bsb[cur][(wn * 64 + j * 16 + l16) * LDP + ks * 32 + 8 * half]);
#pragma unroll
            for (int i = 0; i < 2; ++i)
#pragma unroll
                for (int j = 0; j < 4; ++j)
                    acc[i][j] = __builtin_amdgcn_wmma_f32_16x16x32_bf16(
                        false, afrag[i], false, bfrag[j],
                        (short)0, acc[i][j], false, false);
        }

        wait_async0();        // our async writes into the other buffer done
        __syncthreads();      // everyone's done (reads of cur, writes of other)
        cur ^= 1;
    }

    // epilogue: lane holds col l16 of each tile, rows r + 8*half
#pragma unroll
    for (int i = 0; i < 2; ++i) {
        int gm0 = bm + wm * 32 + i * 16 + half * 8;
#pragma unroll
        for (int j = 0; j < 4; ++j) {
            int gn = bn + wn * 64 + j * 16 + l16;
            float bv = bias[gn];
#pragma unroll
            for (int r = 0; r < 8; ++r) {
                float v = acc[i][j][r] + bv;
                if (relu) v = fmaxf(v, 0.0f);
                size_t o = (size_t)(gm0 + r) * Nc + gn;
                if (Df)  Df[o]  = v;
                if (Dbf) Dbf[o] = (__bf16)v;
            }
        }
    }
}

// ---------------------------------------------------------------------------
// Weight prep: W[K,Nc] f32 -> Wt[Nc,K] bf16 (batched over blockIdx.z)
// ---------------------------------------------------------------------------
__global__ __launch_bounds__(256)
void transpose_cvt_kernel(const float* __restrict__ W, __bf16* __restrict__ Wt,
                          int K, int Nc, size_t wstride, size_t tstride)
{
    __shared__ float tile[32][33];
    const float* Wz  = W  + wstride * blockIdx.z;
    __bf16*      Wtz = Wt + tstride * blockIdx.z;
    int k0 = blockIdx.x * 32, n0 = blockIdx.y * 32;
    int tx = threadIdx.x & 31, ty = threadIdx.x >> 5;   // 32 x 8
#pragma unroll
    for (int i = 0; i < 4; ++i) {
        int r = ty + i * 8;
        tile[r][tx] = Wz[(size_t)(k0 + r) * Nc + n0 + tx];
    }
    __syncthreads();
#pragma unroll
    for (int i = 0; i < 4; ++i) {
        int r = ty + i * 8;
        Wtz[(size_t)(n0 + r) * K + k0 + tx] = (__bf16)tile[tx][r];
    }
}

// ---------------------------------------------------------------------------
// Elementwise f32 -> bf16 (float4 -> 4 bf16 packed as 8B)
// ---------------------------------------------------------------------------
__global__ __launch_bounds__(256)
void cvt_bf16_kernel(const float* __restrict__ in, __bf16* __restrict__ out)
{
    int i = blockIdx.x * blockDim.x + threadIdx.x;
    float4 v = reinterpret_cast<const float4*>(in)[i];
    union { __bf16 h[4]; uint2 u; } p;
    p.h[0] = (__bf16)v.x; p.h[1] = (__bf16)v.y;
    p.h[2] = (__bf16)v.z; p.h[3] = (__bf16)v.w;
    reinterpret_cast<uint2*>(out)[i] = p.u;
}

// ---------------------------------------------------------------------------
// Gating: one wave per (t, n); top-3-of-6 softmax + atomic stats.
// ---------------------------------------------------------------------------
__global__ __launch_bounds__(256)
void gate_kernel(const float* __restrict__ x,
                 const float* __restrict__ w_gate,
                 float* __restrict__ gates,
                 float* __restrict__ imp,
                 float* __restrict__ loadc)
{
    int gw   = (blockIdx.x * blockDim.x + threadIdx.x) >> 5;  // 0 .. T*N-1
    int lane = threadIdx.x & 31;
    int t = gw / N_;
    int n = gw % N_;

    float lg[E_] = {};
    const float* xr = x + (size_t)n * H_;
    const float* wg = w_gate + (size_t)t * H_ * E_;
    for (int h = lane; h < H_; h += 32) {
        float xv = xr[h];
        const float* w = wg + h * E_;
#pragma unroll
        for (int e = 0; e < E_; ++e) lg[e] += xv * w[e];
    }
#pragma unroll
    for (int e = 0; e < E_; ++e)
#pragma unroll
        for (int off = 16; off > 0; off >>= 1)
            lg[e] += __shfl_xor(lg[e], off, 32);

    if (lane == 0) {
        int used = 0, idx[T_];
        float val[T_];
#pragma unroll
        for (int k = 0; k < T_; ++k) {       // top-3 of 6, first-index tiebreak
            float best = -3.0e38f; int bi = 0;
#pragma unroll
            for (int e = 0; e < E_; ++e)
                if (!((used >> e) & 1) && lg[e] > best) { best = lg[e]; bi = e; }
            used |= 1 << bi; val[k] = best; idx[k] = bi;
        }
        float mx = val[0];
        float ex[T_], s = 0.0f;
#pragma unroll
        for (int k = 0; k < T_; ++k) { ex[k] = __expf(val[k] - mx); s += ex[k]; }
        float inv = 1.0f / s;
        float g[E_] = {};
#pragma unroll
        for (int k = 0; k < T_; ++k) g[idx[k]] = ex[k] * inv;

        float* gr = gates + (size_t)(t * N_ + n) * E_;
#pragma unroll
        for (int e = 0; e < E_; ++e) gr[e] = g[e];
#pragma unroll
        for (int k = 0; k < T_; ++k) {
            atomicAdd(&imp[t * E_ + idx[k]], ex[k] * inv);
            atomicAdd(&loadc[t * E_ + idx[k]], 1.0f);
        }
    }
}

// ---------------------------------------------------------------------------
// Per-expert tower dots: dtw[t][n][e] = eo_e[n,:] . tower_w[t,:]
// ---------------------------------------------------------------------------
__global__ __launch_bounds__(256)
void tower_dot_kernel(const float* __restrict__ eo,
                      const float* __restrict__ tower_w,
                      float* __restrict__ dtw, int e)
{
    int n    = (blockIdx.x * blockDim.x + threadIdx.x) >> 5;
    int lane = threadIdx.x & 31;
    float a0 = 0.f, a1 = 0.f, a2 = 0.f;
    const float* row = eo + (size_t)n * H_;
    for (int h = lane; h < H_; h += 32) {
        float v = row[h];
        a0 += v * tower_w[h];
        a1 += v * tower_w[H_ + h];
        a2 += v * tower_w[2 * H_ + h];
    }
#pragma unroll
    for (int off = 16; off > 0; off >>= 1) {
        a0 += __shfl_xor(a0, off, 32);
        a1 += __shfl_xor(a1, off, 32);
        a2 += __shfl_xor(a2, off, 32);
    }
    if (lane == 0) {
        dtw[((size_t)0 * N_ + n) * E_ + e] = a0;
        dtw[((size_t)1 * N_ + n) * E_ + e] = a1;
        dtw[((size_t)2 * N_ + n) * E_ + e] = a2;
    }
}

// ---------------------------------------------------------------------------
// z[t,n] = tower_b[t] + sum_e gates[t,n,e] * dtw[t,n,e]
// ---------------------------------------------------------------------------
__global__ __launch_bounds__(256)
void combine_kernel(const float* __restrict__ gates,
                    const float* __restrict__ dtw,
                    const float* __restrict__ tower_b,
                    float* __restrict__ z)
{
    int i = blockIdx.x * blockDim.x + threadIdx.x;   // 0 .. T*N-1
    int t = i / N_;
    const float* gr = gates + (size_t)i * E_;
    const float* dr = dtw + (size_t)i * E_;
    float s = tower_b[t];
#pragma unroll
    for (int e = 0; e < E_; ++e) s += gr[e] * dr[e];
    z[i] = s;
}

__global__ void colmax_kernel(const float* __restrict__ scores,
                              float* __restrict__ maxsc)
{
    int i = blockIdx.x * blockDim.x + threadIdx.x;
    if (i >= B_ * T_) return;
    int b = i / T_, t = i % T_;
    float m = -3.0e38f;
    for (int c = 0; c < C_; ++c)
        m = fmaxf(m, scores[((size_t)(b * C_ + c)) * T_ + t]);
    maxsc[b * T_ + t] = m;
}

__global__ __launch_bounds__(256)
void bce_preds_kernel(const float* __restrict__ z,
                      const float* __restrict__ scores,
                      const float* __restrict__ maxsc,
                      float* __restrict__ bce_sum,
                      float* __restrict__ preds)
{
    __shared__ float red[256];
    int n = blockIdx.x * blockDim.x + threadIdx.x;   // 0 .. N-1
    int b = n / C_;
    float bsum = 0.f, zsum = 0.f;
#pragma unroll
    for (int t = 0; t < T_; ++t) {
        float zl  = z[(size_t)t * N_ + n];
        float lab = (scores[(size_t)n * T_ + t] == maxsc[b * T_ + t]) ? 1.f : 0.f;
        bsum += fmaxf(zl, 0.f) - zl * lab + log1pf(expf(-fabsf(zl)));
        zsum += zl;
    }
    preds[n] = 1.0f / (1.0f + expf(-zsum / (float)T_));

    red[threadIdx.x] = bsum;
    __syncthreads();
    for (int s = 128; s > 0; s >>= 1) {
        if (threadIdx.x < s) red[threadIdx.x] += red[threadIdx.x + s];
        __syncthreads();
    }
    if (threadIdx.x == 0) atomicAdd(bce_sum, red[0]);
}

__global__ void finalize_kernel(const float* __restrict__ bce_sum,
                                const float* __restrict__ imp,
                                const float* __restrict__ loadc,
                                float* __restrict__ loss_out)
{
    if (threadIdx.x != 0 || blockIdx.x != 0) return;
    float aux = 0.f;
    for (int t = 0; t < T_; ++t) {
        const float* rows[2] = { imp + t * E_, loadc + t * E_ };
        for (int r = 0; r < 2; ++r) {
            float mean = 0.f;
            for (int e = 0; e < E_; ++e) mean += rows[r][e];
            mean /= (float)E_;
            float var = 0.f;
            for (int e = 0; e < E_; ++e) {
                float d = rows[r][e] - mean;
                var += d * d;
            }
            var /= (float)(E_ - 1);                 // ddof=1
            aux += var / (mean * mean + 1e-10f);
        }
    }
    float bce = bce_sum[0] / (float)(B_ * C_ * T_);
    loss_out[0] = bce + 1e-2f * aux;
}

// ---------------------------------------------------------------------------
// Host-side launcher
// ---------------------------------------------------------------------------
extern "C" void kernel_launch(void* const* d_in, const int* in_sizes, int n_in,
                              void* d_out, int out_size, void* d_ws, size_t ws_size,
                              hipStream_t stream)
{
    const float* encs      = (const float*)d_in[0];   // [N,H]
    const float* scores    = (const float*)d_in[1];   // [B,C,T]
    const float* fc1_w     = (const float*)d_in[2];   // [H,H]
    const float* fc1_b     = (const float*)d_in[3];
    const float* fc2_w     = (const float*)d_in[4];
    const float* fc2_b     = (const float*)d_in[5];
    const float* w_gate    = (const float*)d_in[6];   // [T,H,E]
    const float* expert_w1 = (const float*)d_in[7];   // [E,H,H]
    const float* expert_b1 = (const float*)d_in[8];   // [E,H]
    const float* expert_w2 = (const float*)d_in[9];
    const float* expert_b2 = (const float*)d_in[10];
    const float* tower_w   = (const float*)d_in[11];  // [T,H]
    const float* tower_b   = (const float*)d_in[12];  // [T]

    float* out_loss  = (float*)d_out;
    float* out_preds = (float*)d_out + 1;

    // -------- workspace carve-up (256B aligned chunks) --------
    char* p = (char*)d_ws;
    auto alloc = [&](size_t bytes) {
        char* r = p; p += (bytes + 255) & ~(size_t)255; return r;
    };
    const size_t NH = (size_t)N_ * H_;
    const size_t HH = (size_t)H_ * H_;
    float*  xf      = (float*) alloc(NH * 4);          // shared-bottom out (f32)
    float*  eo      = (float*) alloc(NH * 4);          // expert out (f32, reused)
    __bf16* encs_bf = (__bf16*)alloc(NH * 2);
    __bf16* t0b     = (__bf16*)alloc(NH * 2);          // inter-layer act (bf16)
    __bf16* xb      = (__bf16*)alloc(NH * 2);          // shared-bottom out (bf16)
    __bf16* fc1t    = (__bf16*)alloc(HH * 2);
    __bf16* fc2t    = (__bf16*)alloc(HH * 2);
    __bf16* ew1t    = (__bf16*)alloc((size_t)E_ * HH * 2);
    __bf16* ew2t    = (__bf16*)alloc((size_t)E_ * HH * 2);
    float*  gates   = (float*) alloc((size_t)T_ * N_ * E_ * 4);
    float*  dtw     = (float*) alloc((size_t)T_ * N_ * E_ * 4);
    float*  zbuf    = (float*) alloc((size_t)T_ * N_ * 4);
    float*  maxsc   = (float*) alloc((size_t)B_ * T_ * 4);
    float*  imp     = (float*) alloc((T_ * E_ * 2 + 1) * 4);  // imp|load|bce
    float*  loadc   = imp + T_ * E_;
    float*  bce     = loadc + T_ * E_;

    (void)hipMemsetAsync(imp, 0, (size_t)(T_ * E_ * 2 + 1) * sizeof(float), stream);

    // -------- one-time precision prep --------
    cvt_bf16_kernel<<<(NH / 4) / 256, 256, 0, stream>>>(encs, encs_bf);
    dim3 tg(H_ / 32, H_ / 32, 1);
    dim3 tg6(H_ / 32, H_ / 32, E_);
    transpose_cvt_kernel<<<tg,  256, 0, stream>>>(fc1_w, fc1t, H_, H_, 0, 0);
    transpose_cvt_kernel<<<tg,  256, 0, stream>>>(fc2_w, fc2t, H_, H_, 0, 0);
    transpose_cvt_kernel<<<tg6, 256, 0, stream>>>(expert_w1, ew1t, H_, H_, HH, HH);
    transpose_cvt_kernel<<<tg6, 256, 0, stream>>>(expert_w2, ew2t, H_, H_, HH, HH);

    dim3 gg(N_ / BM, H_ / BN);

    // shared bottom
    gemm_bf16_kernel<<<gg, 256, 0, stream>>>(encs_bf, fc1t, fc1_b,
                                             nullptr, t0b, N_, H_, H_, 1);
    gemm_bf16_kernel<<<gg, 256, 0, stream>>>(t0b, fc2t, fc2_b,
                                             xf, xb, N_, H_, H_, 0);

    gate_kernel<<<(T_ * N_ * 32) / 256, 256, 0, stream>>>(xf, w_gate, gates,
                                                          imp, loadc);

    // experts + fused tower dots (eo reused per expert)
    for (int e = 0; e < E_; ++e) {
        gemm_bf16_kernel<<<gg, 256, 0, stream>>>(
            xb, ew1t + (size_t)e * HH, expert_b1 + (size_t)e * H_,
            nullptr, t0b, N_, H_, H_, 1);
        gemm_bf16_kernel<<<gg, 256, 0, stream>>>(
            t0b, ew2t + (size_t)e * HH, expert_b2 + (size_t)e * H_,
            eo, nullptr, N_, H_, H_, 0);
        tower_dot_kernel<<<(N_ * 32) / 256, 256, 0, stream>>>(eo, tower_w, dtw, e);
    }

    combine_kernel<<<(T_ * N_) / 256, 256, 0, stream>>>(gates, dtw, tower_b, zbuf);
    colmax_kernel<<<(B_ * T_ + 255) / 256, 256, 0, stream>>>(scores, maxsc);
    bce_preds_kernel<<<N_ / 256, 256, 0, stream>>>(zbuf, scores, maxsc, bce, out_preds);
    finalize_kernel<<<1, 32, 0, stream>>>(bce, imp, loadc, out_loss);
}
